// MaskedSelfAttention_68564857914103
// MI455X (gfx1250) — matrix operand order repaired and checked by reading, not compiled, not run on hardware
//
#include <hip/hip_runtime.h>

#define NB   4
#define SEQ  2048
#define DIM  1024
#define DKK  128
#define LDA  1032   // 16 x 1024 bf16 strip, padded: 516 dwords/row -> row m starts at bank 4m

typedef __bf16 bf16;
typedef __attribute__((ext_vector_type(16))) __bf16 v16bf;
typedef __attribute__((ext_vector_type(8)))  __bf16 v8bf;
typedef __attribute__((ext_vector_type(4)))  __bf16 v4bf;
typedef __attribute__((ext_vector_type(8)))  float  v8f;
typedef __attribute__((ext_vector_type(4)))  int    v4i;

#define AS1 __attribute__((address_space(1)))
#define AS3 __attribute__((address_space(3)))

#if defined(__gfx1250__) && __has_builtin(__builtin_amdgcn_global_load_async_to_lds_b128)
#define HAVE_ASYNC_LDS 1
#else
#define HAVE_ASYNC_LDS 0
#endif

static __device__ __forceinline__ v16bf join8(v8bf lo, v8bf hi) {
  v16bf r;
#pragma unroll
  for (int i = 0; i < 8; ++i) { r[i] = lo[i]; r[i + 8] = hi[i]; }
  return r;
}

// ---------------------------------------------------------------- converts --
__global__ __launch_bounds__(256) void cvt_x_kernel(const float* __restrict__ X,
                                                    bf16* __restrict__ Xb) {
  size_t i = ((size_t)blockIdx.x * 256 + threadIdx.x) * 4;
  float4 v = *(const float4*)(X + i);
  v4bf o;
  o[0] = (bf16)v.x; o[1] = (bf16)v.y; o[2] = (bf16)v.z; o[3] = (bf16)v.w;
  *(v4bf*)(Xb + i) = o;
}

__global__ __launch_bounds__(256) void cvt_w_kernel(const float* __restrict__ Wq,
                                                    const float* __restrict__ Wk,
                                                    const float* __restrict__ Wv,
                                                    bf16* __restrict__ Wtq,
                                                    bf16* __restrict__ Wtk,
                                                    bf16* __restrict__ Wtv) {
  const int which = blockIdx.y;
  const float* W = which == 0 ? Wq : which == 1 ? Wk : Wv;
  bf16* Wt = which == 0 ? Wtq : which == 1 ? Wtk : Wtv;
  int idx = blockIdx.x * 256 + threadIdx.x;       // over D*DK = 131072
  int d = idx / DKK, n = idx % DKK;
  Wt[(size_t)n * DIM + d] = (bf16)W[idx];         // N-major for contiguous B-frags
}

// -------------------------------------------------------------- QKV GEMM ----
// grid(512, 3), block 256 (8 waves). Wave w computes the 16x16 tile at
// (rows blockIdx.x*16, cols w*16) of Y = X(8192x1024) @ W(1024x128) + bias.
__global__ __launch_bounds__(256) void qkv_gemm_kernel(
    const bf16* __restrict__ Xb,
    const bf16* __restrict__ Wtq, const bf16* __restrict__ Wtk, const bf16* __restrict__ Wtv,
    const float* __restrict__ bq, const float* __restrict__ bk, const float* __restrict__ bv,
    bf16* __restrict__ Qb, bf16* __restrict__ Kb, bf16* __restrict__ Vtb) {
  __shared__ __align__(16) bf16 As[16 * LDA];

  const int tid  = threadIdx.x;
  const int w    = tid >> 5;
  const int l    = tid & 31;
  const int half = l >> 4;
  const int lm   = l & 15;
  const int row0 = blockIdx.x * 16;
  const int which = blockIdx.y;
  const bf16*  Wt   = which == 0 ? Wtq : which == 1 ? Wtk : Wtv;
  const float* bias = which == 0 ? bq  : which == 1 ? bk  : bv;

  // cooperative staging: 16 rows x 1024 bf16 = 2048 x b128 chunks.
  // CDNA5 async direct-to-LDS path (ASYNCcnt) when the builtin is declared.
#pragma unroll
  for (int it = 0; it < 8; ++it) {
    int ch = tid + it * 256;                 // 0..2047
    int r  = ch >> 7;
    int c8 = (ch & 127) * 8;
    const bf16* src = Xb + (size_t)(row0 + r) * DIM + c8;
    bf16* dst = As + r * LDA + c8;
#if HAVE_ASYNC_LDS
    __builtin_amdgcn_global_load_async_to_lds_b128(
        (AS1 v4i*)(void*)src, (AS3 v4i*)(void*)dst, 0, 0);
#else
    *(v8bf*)dst = *(const v8bf*)src;
#endif
  }
#if HAVE_ASYNC_LDS
  asm volatile("s_wait_asynccnt 0" ::: "memory");
#endif
  __syncthreads();

  const int n0 = w * 16;
  const int n  = n0 + lm;
  v8f c = {};
#pragma unroll 4
  for (int kt = 0; kt < 32; ++kt) {
    const int k0 = kt * 32;
    // A-frag (lanes0-15 hold K=0..15, lanes16-31 K=16..31, row m per lane)
    v8bf alo = *(const v8bf*)(As + lm * LDA + k0 + half * 16);
    v8bf ahi = *(const v8bf*)(As + lm * LDA + k0 + half * 16 + 8);
    v16bf a = join8(alo, ahi);
    // B-frag (lanes0-15 hold K=0..15, lanes16-31 K=16..31, column n per lane)
    const bf16* wp = Wt + (size_t)n * DIM + k0 + half * 16;
    __builtin_prefetch(wp + 32, 0, 3);
    v16bf bfr = join8(*(const v8bf*)wp, *(const v8bf*)(wp + 8));
    c = __builtin_amdgcn_wmma_f32_16x16x32_bf16(false, a, false, bfr, (short)0, c,
                                                false, false);
  }

  const float bias_n = bias[n];
#pragma unroll
  for (int r = 0; r < 8; ++r) {
    const int m   = r + half * 8;
    const int row = row0 + m;
    const float v = c[r] + bias_n;
    if (which == 2) {                         // V stored transposed [B][DK][S]
      const int bb = row >> 11, s = row & (SEQ - 1);
      Vtb[((size_t)bb * DKK + n) * SEQ + s] = (bf16)v;
    } else {
      bf16* Y = which == 0 ? Qb : Kb;         // [B*S][DK] row-major
      Y[(size_t)row * DKK + n] = (bf16)v;
    }
  }
}

// ------------------------------------------------------------- attention ----
// grid(S/16=128, B), block 256 (8 waves). Each block owns a 16-query tile;
// waves split the 64 key super-tiles (32 keys each) with private online
// softmax, then merge through LDS f32 atomics.
__global__ __launch_bounds__(256) void attn_kernel(
    const bf16* __restrict__ Qb, const bf16* __restrict__ Kb,
    const bf16* __restrict__ Vtb, float* __restrict__ Out) {
  __shared__ __align__(16) bf16 Pl[8][16 * 40];   // per-wave P staging (pad 40)
  __shared__ float Oacc[16][DKK];
  __shared__ float Lacc[16];
  __shared__ float Ml[8][16];

  const int tid  = threadIdx.x;
  const int w    = tid >> 5;
  const int l    = tid & 31;
  const int half = l >> 4;
  const int lm   = l & 15;
  const int b    = blockIdx.y;
  const int q0   = blockIdx.x * 16;
  const float scale = 0.08838834764831845f;       // 1/sqrt(128)

  for (int i = tid; i < 16 * DKK; i += 256) ((float*)Oacc)[i] = 0.f;
  if (tid < 16) Lacc[tid] = 0.f;
  __syncthreads();

  // Q A-fragments for rows q0..q0+15, DK slices kt*32
  v16bf aq[4];
#pragma unroll
  for (int kt = 0; kt < 4; ++kt) {
    const bf16* qp = Qb + ((size_t)(b * SEQ + q0 + lm)) * DKK + kt * 32;
    aq[kt] = join8(*(const v8bf*)(qp + half * 16),
                   *(const v8bf*)(qp + half * 16 + 8));
  }

  const v8f vzero = {};
  float m_run[8], l_run[8];
  v8f acc[8];
#pragma unroll
  for (int r = 0; r < 8; ++r) { m_run[r] = -3.0e38f; l_run[r] = 0.f; acc[r] = vzero; }

  for (int kb = w; kb < SEQ / 32; kb += 8) {
    const int key0 = kb * 32;
    v8f c0 = vzero, c1 = vzero;
#pragma unroll
    for (int kt = 0; kt < 4; ++kt) {
      const bf16* kp0 = Kb + ((size_t)(b * SEQ + key0 + lm)) * DKK + kt * 32 + half * 16;
      v16bf b0 = join8(*(const v8bf*)kp0, *(const v8bf*)(kp0 + 8));
      c0 = __builtin_amdgcn_wmma_f32_16x16x32_bf16(false, aq[kt], false, b0, (short)0,
                                                   c0, false, false);
      const bf16* kp1 = kp0 + 16 * DKK;
      v16bf b1 = join8(*(const v8bf*)kp1, *(const v8bf*)(kp1 + 8));
      c1 = __builtin_amdgcn_wmma_f32_16x16x32_bf16(false, aq[kt], false, b1, (short)0,
                                                   c1, false, false);
    }

    // scale + faithful mask (key <= query masked) + online softmax update
    float p0[8], p1[8], fac[8];
#pragma unroll
    for (int r = 0; r < 8; ++r) {
      const int q   = q0 + r + half * 8;
      const int k0i = key0 + lm;
      const int k1i = key0 + 16 + lm;
      float s0 = c0[r] * scale + (k0i <= q ? -1.0e9f : 0.f);
      float s1 = c1[r] * scale + (k1i <= q ? -1.0e9f : 0.f);
      float mx = fmaxf(s0, s1);
      mx = fmaxf(mx, __shfl_xor(mx, 1, 32));
      mx = fmaxf(mx, __shfl_xor(mx, 2, 32));
      mx = fmaxf(mx, __shfl_xor(mx, 4, 32));
      mx = fmaxf(mx, __shfl_xor(mx, 8, 32));
      const float mn = fmaxf(m_run[r], mx);
      fac[r] = __expf(m_run[r] - mn);
      p0[r]  = __expf(s0 - mn);
      p1[r]  = __expf(s1 - mn);
      float rs = p0[r] + p1[r];
      rs += __shfl_xor(rs, 1, 32);
      rs += __shfl_xor(rs, 2, 32);
      rs += __shfl_xor(rs, 4, 32);
      rs += __shfl_xor(rs, 8, 32);
      l_run[r] = l_run[r] * fac[r] + rs;
      m_run[r] = mn;
    }
#pragma unroll
    for (int t = 0; t < 8; ++t)
#pragma unroll
      for (int r = 0; r < 8; ++r) acc[t][r] *= fac[r];

    // stage P (C-layout) through per-wave LDS, reload as A-fragment
    bf16* pw = Pl[w];
#pragma unroll
    for (int r = 0; r < 8; ++r) {
      const int m = r + half * 8;
      pw[m * 40 + lm]      = (bf16)p0[r];
      pw[m * 40 + 16 + lm] = (bf16)p1[r];
    }
    asm volatile("s_wait_dscnt 0" ::: "memory");
    v16bf ap = join8(*(const v8bf*)(pw + lm * 40 + half * 16),
                     *(const v8bf*)(pw + lm * 40 + half * 16 + 8));

    // O += P @ V over all 8 N-tiles (V transposed -> contiguous B-frags)
#pragma unroll
    for (int t = 0; t < 8; ++t) {
      const bf16* vp = Vtb + ((size_t)b * DKK + t * 16 + lm) * SEQ + key0 + half * 16;
      v16bf bfr = join8(*(const v8bf*)vp, *(const v8bf*)(vp + 8));
      acc[t] = __builtin_amdgcn_wmma_f32_16x16x32_bf16(false, ap, false, bfr, (short)0,
                                                       acc[t], false, false);
    }
  }

  // cross-wave merge
  if (lm == 0) {
#pragma unroll
    for (int r = 0; r < 8; ++r) Ml[w][r + half * 8] = m_run[r];
  }
  __syncthreads();

  float fac2[8];
#pragma unroll
  for (int r = 0; r < 8; ++r) {
    const int m = r + half * 8;
    float gm = Ml[0][m];
#pragma unroll
    for (int ww = 1; ww < 8; ++ww) gm = fmaxf(gm, Ml[ww][m]);
    fac2[r] = __expf(m_run[r] - gm);
  }
  if (lm == 0) {
#pragma unroll
    for (int r = 0; r < 8; ++r) atomicAdd(&Lacc[r + half * 8], l_run[r] * fac2[r]);
  }
#pragma unroll
  for (int t = 0; t < 8; ++t)
#pragma unroll
    for (int r = 0; r < 8; ++r)
      atomicAdd(&Oacc[r + half * 8][t * 16 + lm], acc[t][r] * fac2[r]);
  __syncthreads();

  for (int i = tid; i < 16 * DKK; i += 256) {
    const int m = i >> 7, n = i & 127;
    Out[((size_t)(b * SEQ + q0 + m)) * DKK + n] = Oacc[m][n] / Lacc[m];
  }
}

// ------------------------------------------------------------------ launch --
extern "C" void kernel_launch(void* const* d_in, const int* in_sizes, int n_in,
                              void* d_out, int out_size, void* d_ws, size_t ws_size,
                              hipStream_t stream) {
  const float* X  = (const float*)d_in[0];
  const float* Wq = (const float*)d_in[1];
  const float* bq = (const float*)d_in[2];
  const float* Wk = (const float*)d_in[3];
  const float* bk = (const float*)d_in[4];
  const float* Wv = (const float*)d_in[5];
  const float* bv = (const float*)d_in[6];
  float* Out = (float*)d_out;

  char* ws = (char*)d_ws;
  size_t off = 0;
  bf16* Xb  = (bf16*)(ws + off); off += (size_t)NB * SEQ * DIM * 2;  // 16 MB
  bf16* Wtq = (bf16*)(ws + off); off += (size_t)DIM * DKK * 2;
  bf16* Wtk = (bf16*)(ws + off); off += (size_t)DIM * DKK * 2;
  bf16* Wtv = (bf16*)(ws + off); off += (size_t)DIM * DKK * 2;
  bf16* Qb  = (bf16*)(ws + off); off += (size_t)NB * SEQ * DKK * 2;
  bf16* Kb  = (bf16*)(ws + off); off += (size_t)NB * SEQ * DKK * 2;
  bf16* Vtb = (bf16*)(ws + off); off += (size_t)NB * SEQ * DKK * 2;
  (void)ws_size; (void)in_sizes; (void)n_in; (void)out_size;

  // 1) input f32 -> bf16 (4 elems/thread)
  cvt_x_kernel<<<(NB * SEQ * DIM) / (256 * 4), 256, 0, stream>>>(X, Xb);
  // 2) weights f32 -> bf16, transposed to N-major
  cvt_w_kernel<<<dim3((DIM * DKK) / 256, 3), 256, 0, stream>>>(Wq, Wk, Wv, Wtq, Wtk, Wtv);
  // 3) Q/K/V projections (bf16 WMMA, f32 accum)
  qkv_gemm_kernel<<<dim3((NB * SEQ) / 16, 3), 256, 0, stream>>>(
      Xb, Wtq, Wtk, Wtv, bq, bk, bv, Qb, Kb, Vtb);
  // 4) masked attention (flash-style, bf16 WMMA)
  attn_kernel<<<dim3(SEQ / 16, NB), 256, 0, stream>>>(Qb, Kb, Vtb, Out);
}
